// DeepAveragingLSTMNetwork_84335977824665
// MI455X (gfx1250) — matrix operand order repaired and checked by reading, not compiled, not run on hardware
//
#include <hip/hip_runtime.h>
#include <hip/hip_bf16.h>

typedef _Float16 v16h __attribute__((ext_vector_type(16)));
typedef _Float16 h8   __attribute__((ext_vector_type(8)));
typedef float    v8f  __attribute__((ext_vector_type(8)));

#define N_WORDS   4096
#define WORD_LEN  16
#define CHAR_EMB  50
#define CHAR_HID  128
#define GATES     512          // 4*CHAR_HID
#define KPAD      192          // 64 (input, padded from 50) + 128 (hidden)
#define GLOVE_DIM 300
#define HIDDEN    512
#define FEAT      428          // GLOVE_DIM + CHAR_HID
#define BFRAG_ELEMS (32*6*32*16)   // colTiles * kChunks * lanes * halves = 98304

// ---------------------------------------------------------------------------
// Kernel 1: zero accumulators, fuse biases
// ---------------------------------------------------------------------------
__global__ void prep_misc_kernel(const float* __restrict__ b_ih,
                                 const float* __restrict__ b_hh,
                                 float* __restrict__ sum_glove,
                                 float* __restrict__ sum_h,
                                 float* __restrict__ bias) {
  int t = threadIdx.x;                       // 512 threads, 1 block
  if (t < GLOVE_DIM) sum_glove[t] = 0.f;
  if (t < CHAR_HID)  sum_h[t]     = 0.f;
  bias[t] = b_ih[t] + b_hh[t];
}

// ---------------------------------------------------------------------------
// Kernel 2: build f16 B-matrix in WMMA fragment order.
// B[k, j]: k in [0,50) -> W_ih[j,k]; k in [64,192) -> W_hh[j,k-64]; else 0.
// Fragment layout per 32x16 B tile (contiguous k order used by this code, matching A):
//   lane 0-15  : N = lane,    halves h=0..15 -> K = h
//   lane 16-31 : N = lane-16, halves h=0..15 -> K = 16 + h
// Stored flat: Bfrag[(((ct*6 + kb)*32 + lane)*16) + h]
// ---------------------------------------------------------------------------
__global__ void prep_bfrag_kernel(const float* __restrict__ W_ih,
                                  const float* __restrict__ W_hh,
                                  _Float16* __restrict__ Bfrag) {
  int fi = blockIdx.x * 256 + threadIdx.x;
  if (fi >= BFRAG_ELEMS) return;
  int h    = fi & 15;
  int lane = (fi >> 4) & 31;
  int ctkb = fi >> 9;
  int kb   = ctkb % 6;
  int ct   = ctkb / 6;
  int k    = kb * 32 + h + 16 * (lane >> 4);
  int col  = ct * 16 + (lane & 15);
  float v = 0.f;
  if (k < 64) {
    if (k < CHAR_EMB) v = W_ih[col * CHAR_EMB + k];
  } else {
    v = W_hh[col * CHAR_HID + (k - 64)];
  }
  Bfrag[fi] = (_Float16)v;
}

// ---------------------------------------------------------------------------
// Kernel 3: sum of gathered glove rows (only the mean is needed)
// ---------------------------------------------------------------------------
__global__ void __launch_bounds__(256)
glove_sum_kernel(const int* __restrict__ word_indices,
                 const float* __restrict__ glove_table,
                 float* __restrict__ sum_glove) {
  __shared__ int wi[16];
  int tid = threadIdx.x;
  if (tid < 16) wi[tid] = word_indices[blockIdx.x * 16 + tid];
  __syncthreads();
  for (int d = tid; d < GLOVE_DIM; d += 256) {
    float s = 0.f;
    #pragma unroll
    for (int w = 0; w < 16; ++w)
      s += glove_table[(size_t)wi[w] * GLOVE_DIM + d];
    atomicAdd(&sum_glove[d], s);
  }
}

// ---------------------------------------------------------------------------
// Kernel 4: char-LSTM via WMMA. One block = 16 words, 256 threads = 8 waves.
// Wave w owns gate columns [w*64, w*64+64) as four 16x16 tiles.
// A (16 x KPAD) staged in LDS as f16: cols 0..63 = x_t (padded), 64..191 = h.
// ---------------------------------------------------------------------------
__global__ void __launch_bounds__(256)
lstm_kernel(const int* __restrict__ char_indices,
            const float* __restrict__ char_embed,
            const _Float16* __restrict__ Bfrag,
            const float* __restrict__ bias,
            float* __restrict__ sum_h) {
  __shared__ __attribute__((aligned(16))) _Float16 Ah[16 * KPAD];  // 6 KB
  __shared__ float gates[16 * GATES];                              // 32 KB
  __shared__ float cst[16 * CHAR_HID];                             // 8 KB
  __shared__ float hst[16 * CHAR_HID];                             // 8 KB

  const int tid   = threadIdx.x;
  const int lane  = tid & 31;
  const int wv    = tid >> 5;        // wave 0..7
  const int grp   = lane >> 4;       // lane half-group
  const int ln16  = lane & 15;
  const int word0 = blockIdx.x * 16;

  // init state and hidden part of A
  for (int e = tid; e < 16 * CHAR_HID; e += 256) {
    cst[e] = 0.f;
    hst[e] = 0.f;
    int w = e >> 7, j = e & 127;
    Ah[w * KPAD + 64 + j] = (_Float16)0.f;
  }
  __syncthreads();

  for (int t = 0; t < WORD_LEN; ++t) {
    // ---- stage x_t (char embedding gather), pad 50 -> 64 ----
    for (int e = tid; e < 16 * 64; e += 256) {
      int w = e >> 6, k = e & 63;
      float v = 0.f;
      if (k < CHAR_EMB) {
        int ci = char_indices[(word0 + w) * WORD_LEN + t];
        v = char_embed[ci * CHAR_EMB + k];
      }
      Ah[w * KPAD + k] = (_Float16)v;
    }
    __syncthreads();

    // ---- load 6 A-fragments (shared across this wave's 4 column tiles) ----
    // A 16x32 layout used here (contiguous k, same as Bfrag): lane<16 -> K {kb*32+0..15},
    //                        lane>=16 -> K {kb*32+16..31}; M = lane&15
    v16h afr[6];
    #pragma unroll
    for (int kb = 0; kb < 6; ++kb) {
      h8 x0 = *(const h8*)&Ah[ln16 * KPAD + kb * 32 + 16 * grp];
      h8 x1 = *(const h8*)&Ah[ln16 * KPAD + kb * 32 + 16 * grp + 8];
      v16h a;
      #pragma unroll
      for (int i = 0; i < 8; ++i) { a[i] = x0[i]; a[8 + i] = x1[i]; }
      afr[kb] = a;
    }

    // ---- GEMM: gates = A * B + bias ----
    #pragma unroll
    for (int ctl = 0; ctl < 4; ++ctl) {
      const int ctIdx = wv * 4 + ctl;
      const int col0  = ctIdx * 16;
      const float bv  = bias[col0 + ln16];   // bias is per-column (N = lane&15)
      v8f acc;
      #pragma unroll
      for (int r = 0; r < 8; ++r) acc[r] = bv;
      #pragma unroll
      for (int kb = 0; kb < 6; ++kb) {
        v16h b = *(const v16h*)&Bfrag[(size_t)(((ctIdx * 6 + kb) * 32 + lane)) * 16];
        acc = __builtin_amdgcn_wmma_f32_16x16x32_f16(
            /*neg_a=*/false, afr[kb], /*neg_b=*/false, b,
            /*c_mod=*/(short)0, acc, /*reuse_a=*/false, /*reuse_b=*/false);
      }
      // D layout: VGPR r -> M = r + 8*grp, N = ln16
      #pragma unroll
      for (int r = 0; r < 8; ++r)
        gates[(r + 8 * grp) * GATES + col0 + ln16] = acc[r];
    }
    __syncthreads();

    // ---- LSTM cell update ----
    for (int e = tid; e < 16 * CHAR_HID; e += 256) {
      int w = e >> 7, j = e & 127;
      float ig = gates[w * GATES + j];
      float fg = gates[w * GATES + 128 + j];
      float gg = gates[w * GATES + 256 + j];
      float og = gates[w * GATES + 384 + j];
      float si = 1.f / (1.f + __expf(-ig));
      float sf = 1.f / (1.f + __expf(-fg));
      float so = 1.f / (1.f + __expf(-og));
      float c  = sf * cst[e] + si * tanhf(gg);
      float h  = so * tanhf(c);
      cst[e] = c;
      hst[e] = h;
      Ah[w * KPAD + 64 + j] = (_Float16)h;   // next timestep's A hidden part
    }
    __syncthreads();
  }

  // ---- reduce final h over this block's 16 words ----
  if (tid < CHAR_HID) {
    float s = 0.f;
    #pragma unroll
    for (int w = 0; w < 16; ++w) s += hst[w * CHAR_HID + tid];
    atomicAdd(&sum_h[tid], s);
  }
}

// ---------------------------------------------------------------------------
// Kernel 5: avg -> fc1+ReLU -> fc2 (single row; trivial)
// ---------------------------------------------------------------------------
__global__ void __launch_bounds__(512)
head_kernel(const float* __restrict__ sum_glove,
            const float* __restrict__ sum_h,
            const float* __restrict__ fc1_w, const float* __restrict__ fc1_b,
            const float* __restrict__ fc2_w, const float* __restrict__ fc2_b,
            float* __restrict__ out) {
  __shared__ float avg[FEAT];
  __shared__ float hid[HIDDEN];
  int t = threadIdx.x;
  const float inv = 1.f / (float)N_WORDS;
  if (t < GLOVE_DIM)      avg[t] = sum_glove[t] * inv;
  else if (t < FEAT)      avg[t] = sum_h[t - GLOVE_DIM] * inv;
  __syncthreads();
  float a = fc1_b[t];
  for (int k = 0; k < FEAT; ++k) a += avg[k] * fc1_w[t * FEAT + k];
  hid[t] = a > 0.f ? a : 0.f;
  __syncthreads();
  if (t < 2) {
    float o = fc2_b[t];
    for (int k = 0; k < HIDDEN; ++k) o += hid[k] * fc2_w[t * HIDDEN + k];
    out[t] = o;
  }
}

// ---------------------------------------------------------------------------
// Launch
// ---------------------------------------------------------------------------
extern "C" void kernel_launch(void* const* d_in, const int* in_sizes, int n_in,
                              void* d_out, int out_size, void* d_ws, size_t ws_size,
                              hipStream_t stream) {
  const int*   word_indices = (const int*)  d_in[0];
  const int*   char_indices = (const int*)  d_in[1];
  const float* glove_table  = (const float*)d_in[2];
  const float* char_embed   = (const float*)d_in[3];
  const float* W_ih         = (const float*)d_in[4];
  const float* W_hh         = (const float*)d_in[5];
  const float* b_ih         = (const float*)d_in[6];
  const float* b_hh         = (const float*)d_in[7];
  const float* fc1_w        = (const float*)d_in[8];
  const float* fc1_b        = (const float*)d_in[9];
  const float* fc2_w        = (const float*)d_in[10];
  const float* fc2_b        = (const float*)d_in[11];
  float* out = (float*)d_out;

  // workspace layout (bytes): needs ~205 KB
  char* ws = (char*)d_ws;
  float*    sum_glove = (float*)(ws + 0);        // 300 f32
  float*    sum_h     = (float*)(ws + 2048);     // 128 f32
  float*    bias      = (float*)(ws + 4096);     // 512 f32
  _Float16* Bfrag     = (_Float16*)(ws + 8192);  // 98304 f16 = 192 KB

  prep_misc_kernel <<<1, 512, 0, stream>>>(b_ih, b_hh, sum_glove, sum_h, bias);
  prep_bfrag_kernel<<<(BFRAG_ELEMS + 255) / 256, 256, 0, stream>>>(W_ih, W_hh, Bfrag);
  glove_sum_kernel <<<N_WORDS / 16, 256, 0, stream>>>(word_indices, glove_table, sum_glove);
  lstm_kernel      <<<N_WORDS / 16, 256, 0, stream>>>(char_indices, char_embed, Bfrag, bias, sum_h);
  head_kernel      <<<1, 512, 0, stream>>>(sum_glove, sum_h, fc1_w, fc1_b, fc2_w, fc2_b, out);
}
